// TransformerEncoder_7258494730495
// MI455X (gfx1250) — matrix operand order repaired and checked by reading, not compiled, not run on hardware
//
#include <hip/hip_runtime.h>
#include <hip/hip_bf16.h>

// ---------------------------------------------------------------------------
// Types for CDNA5 WMMA (wave32, 16x16x32 bf16 -> f32 accumulate)
// ---------------------------------------------------------------------------
typedef __attribute__((ext_vector_type(16))) __bf16 v16bf;
typedef __attribute__((ext_vector_type(8)))  float  v8f;

union AFrag { v16bf v; unsigned short u[16]; };

__device__ __forceinline__ unsigned short f2bf(float f) {
    union { float f; unsigned u; } a; a.f = f;
    unsigned r = a.u + 0x7FFFu + ((a.u >> 16) & 1u);   // round-to-nearest-even
    return (unsigned short)(r >> 16);
}

// ---------------------------------------------------------------------------
// bf16 WMMA GEMM with async LDS staging:
//   C[M,N] = act( A[M,K] @ Bt[N,K]^T + bias + res1 + res2 )
// A is bf16 row-major [M,K]; Bt is bf16 PRE-TRANSPOSED weights [N,K].
// Both tiles are staged to LDS with GLOBAL_LOAD_ASYNC_TO_LDS_B128 (ASYNCcnt),
// so the inner loop has zero conversion VALU.
// Block tile 128x128x32, 256 threads = 8 wave32s, wave sub-tile 32x64
// = 2x4 WMMA fragments -> 8 WMMAs / K-step.
// ---------------------------------------------------------------------------
#define BM 128
#define BN 128
#define BK 32
#define ROWE 40              // LDS row stride in bf16 elems (80 B, 16B-aligned)

__global__ __launch_bounds__(256) void gemm_bf16(
    const unsigned short* __restrict__ A, const unsigned short* __restrict__ Bt,
    const float* __restrict__ bias, const float* __restrict__ res1,
    const float* __restrict__ res2, float* __restrict__ C,
    unsigned short* __restrict__ Cb,
    int M, int N, int K, int act)
{
    __shared__ __align__(16) unsigned short As[BM * ROWE];
    __shared__ __align__(16) unsigned short Bs[BN * ROWE];

    const int tid  = threadIdx.x;
    const int n0   = blockIdx.x * BN;
    const int m0   = blockIdx.y * BM;
    const int wv   = tid >> 5;
    const int lane = tid & 31;
    const int half = lane >> 4;
    const int lrow = lane & 15;
    const int wm   = (wv & 3) * 32;   // wave M offset inside block tile
    const int wn   = (wv >> 2) * 64;  // wave N offset inside block tile

    const unsigned ldsA0 = (unsigned)(size_t)(void*)As;  // low 32 bits = LDS addr
    const unsigned ldsB0 = (unsigned)(size_t)(void*)Bs;

    v8f acc[2][4] = {};

    for (int k0 = 0; k0 < K; k0 += BK) {
        // ---- async-stage A tile: 128 rows x 32 bf16 = 512 x 16B chunks ----
        #pragma unroll
        for (int it = 0; it < 2; ++it) {
            int i  = tid + it * 256;
            int r  = i >> 2;             // tile row
            int ce = (i & 3) * 8;        // element offset in row (16B chunk)
            int gr = m0 + r;
            unsigned lds = ldsA0 + (unsigned)(r * ROWE + ce) * 2;
            if (gr < M) {
                unsigned long long ga =
                    (unsigned long long)(size_t)(A + (size_t)gr * K + k0 + ce);
                asm volatile("global_load_async_to_lds_b128 %0, %1, off"
                             :: "v"(lds), "v"(ga) : "memory");
            } else {
                *(uint4*)(As + r * ROWE + ce) = make_uint4(0u, 0u, 0u, 0u);
            }
        }
        // ---- async-stage B tile: 128 rows (n) x 32 bf16 (k) ----
        #pragma unroll
        for (int it = 0; it < 2; ++it) {
            int i  = tid + it * 256;
            int r  = i >> 2;
            int ce = (i & 3) * 8;
            unsigned lds = ldsB0 + (unsigned)(r * ROWE + ce) * 2;
            unsigned long long ga =
                (unsigned long long)(size_t)(Bt + (size_t)(n0 + r) * K + k0 + ce);
            asm volatile("global_load_async_to_lds_b128 %0, %1, off"
                         :: "v"(lds), "v"(ga) : "memory");
        }
        // ---- prefetch next K-tiles toward L2 while DMA runs ----
        if (k0 + BK < K) {
            int r = tid >> 2, ce = (tid & 3) * 8;
            int gr = m0 + r;
            if (gr < M) __builtin_prefetch(A + (size_t)gr * K + (k0 + BK) + ce, 0, 1);
            __builtin_prefetch(Bt + (size_t)(n0 + r) * K + (k0 + BK) + ce, 0, 1);
        }
        asm volatile("s_wait_asynccnt 0x0" ::: "memory");
        __syncthreads();

        // ---- build fragments per ISA 16-bit A(16x32) / B(32x16) layouts ----
        AFrag a[2], b[4];
        #pragma unroll
        for (int tm = 0; tm < 2; ++tm) {
            const unsigned short* ap = As + (wm + tm * 16 + lrow) * ROWE;
            #pragma unroll
            for (int e = 0; e < 8; ++e) a[tm].u[e]     = ap[half * 8 + e];
            #pragma unroll
            for (int e = 0; e < 8; ++e) a[tm].u[8 + e] = ap[16 + half * 8 + e];
        }
        #pragma unroll
        for (int tn = 0; tn < 4; ++tn) {
            const unsigned short* bp = Bs + (wn + tn * 16 + lrow) * ROWE;
            #pragma unroll
            for (int e = 0; e < 8; ++e) b[tn].u[e]     = bp[half * 8 + e];
            #pragma unroll
            for (int e = 0; e < 8; ++e) b[tn].u[8 + e] = bp[16 + half * 8 + e];
        }
        #pragma unroll
        for (int tm = 0; tm < 2; ++tm)
            #pragma unroll
            for (int tn = 0; tn < 4; ++tn)
                acc[tm][tn] = __builtin_amdgcn_wmma_f32_16x16x32_bf16(
                    false, a[tm].v, false, b[tn].v, (short)0, acc[tm][tn],
                    false, false);
        __syncthreads();
    }

    // ---- epilogue: bias + residual(s) + optional exact GELU ----
    #pragma unroll
    for (int tm = 0; tm < 2; ++tm) {
        #pragma unroll
        for (int tn = 0; tn < 4; ++tn) {
            int col = n0 + wn + tn * 16 + lrow;
            #pragma unroll
            for (int r = 0; r < 8; ++r) {
                int row = m0 + wm + tm * 16 + half * 8 + r;
                if (row < M) {
                    float v = acc[tm][tn][r];
                    if (bias) v += bias[col];
                    size_t idx = (size_t)row * N + col;
                    if (res1) v += res1[idx];
                    if (res2) v += res2[idx];
                    if (act)  v = 0.5f * v * (1.0f + erff(v * 0.70710678118f));
                    if (C)  C[idx]  = v;
                    if (Cb) Cb[idx] = f2bf(v);
                }
            }
        }
    }
}

// ---------------------------------------------------------------------------
// Weight pack: [K,N] f32 -> [N,K] bf16 (transpose + convert), one-time.
// ---------------------------------------------------------------------------
__global__ __launch_bounds__(256) void pack_tr(
    const float* __restrict__ src, unsigned short* __restrict__ dst, int K, int N)
{
    int i = blockIdx.x * 256 + threadIdx.x;
    if (i >= K * N) return;
    int n = i / K, k = i - n * K;
    dst[i] = f2bf(src[(size_t)k * N + n]);
}

// ---------------------------------------------------------------------------
// LayerNorm over D=512, one block per row. Emits f32 (residual path) and
// bf16 (WMMA A-operand path).
// ---------------------------------------------------------------------------
__global__ __launch_bounds__(256) void ln_kernel(
    const float* __restrict__ x, const float* __restrict__ g,
    const float* __restrict__ b, float* __restrict__ y,
    unsigned short* __restrict__ yb)
{
    __shared__ float red[256];
    const int tid = threadIdx.x;
    const size_t row = blockIdx.x;
    const float* xr = x + row * 512;
    float v0 = xr[tid], v1 = xr[tid + 256];

    red[tid] = v0 + v1;
    __syncthreads();
    for (int o = 128; o > 0; o >>= 1) {
        if (tid < o) red[tid] += red[tid + o];
        __syncthreads();
    }
    float mu = red[0] * (1.0f / 512.0f);
    __syncthreads();

    float d0 = v0 - mu, d1 = v1 - mu;
    red[tid] = d0 * d0 + d1 * d1;
    __syncthreads();
    for (int o = 128; o > 0; o >>= 1) {
        if (tid < o) red[tid] += red[tid + o];
        __syncthreads();
    }
    float rs = rsqrtf(red[0] * (1.0f / 512.0f) + 1e-5f);

    float o0 = d0 * rs * g[tid]       + b[tid];
    float o1 = d1 * rs * g[tid + 256] + b[tid + 256];
    float* yr = y + row * 512;
    yr[tid] = o0; yr[tid + 256] = o1;
    unsigned short* ybr = yb + row * 512;
    ybr[tid] = f2bf(o0); ybr[tid + 256] = f2bf(o1);
}

// ---------------------------------------------------------------------------
// [B, d0, d1, 512] -> [B, d1, d0, 512] transpose (float4 granularity)
// ---------------------------------------------------------------------------
__global__ __launch_bounds__(256) void transpose_kernel(
    const float4* __restrict__ in, float4* __restrict__ out,
    int d0, int d1, size_t total)
{
    size_t i = (size_t)blockIdx.x * blockDim.x + threadIdx.x;
    if (i >= total) return;
    int c = (int)(i & 127);
    size_t j = i >> 7;
    int n1 = (int)(j % d1); j /= d1;
    int n0 = (int)(j % d0);
    size_t bb = j / d0;
    out[(((bb * d1) + n1) * d0 + n0) * 128 + c] = in[i];
}

// ---------------------------------------------------------------------------
// Spatial attention: S=25, one block per (batch, head). K/V/P in LDS.
// qkv rows have stride 1536 (q | k | v each 512 = 8 heads x 64).
// Output bf16 (consumed only as GEMM A-operand).
// ---------------------------------------------------------------------------
#define SS 25
__global__ __launch_bounds__(256) void attn_spatial(
    const float* __restrict__ qkv, unsigned short* __restrict__ out)
{
    __shared__ float Ks[SS * 64], Vs[SS * 64], P[SS * SS];
    const int tid = threadIdx.x;
    const int bb  = blockIdx.x >> 3;
    const int h   = blockIdx.x & 7;
    const float* base = qkv + (size_t)bb * SS * 1536 + h * 64;

    for (int i = tid; i < SS * 64; i += 256) {
        int s = i >> 6, d = i & 63;
        Ks[i] = base[(size_t)s * 1536 + 512 + d];
        Vs[i] = base[(size_t)s * 1536 + 1024 + d];
    }
    __syncthreads();

    for (int i = tid; i < SS * SS; i += 256) {
        int qi = i / SS, kj = i % SS;
        const float* qp = base + (size_t)qi * 1536;
        float s = 0.f;
        #pragma unroll 16
        for (int d = 0; d < 64; ++d) s += qp[d] * Ks[kj * 64 + d];
        P[i] = s * 0.125f;  // 1/sqrt(64)
    }
    __syncthreads();

    if (tid < SS) {
        float m = -1e30f;
        for (int j = 0; j < SS; ++j) m = fmaxf(m, P[tid * SS + j]);
        float l = 0.f;
        for (int j = 0; j < SS; ++j) {
            float e = __expf(P[tid * SS + j] - m);
            P[tid * SS + j] = e; l += e;
        }
        float inv = 1.f / l;
        for (int j = 0; j < SS; ++j) P[tid * SS + j] *= inv;
    }
    __syncthreads();

    for (int i = tid; i < SS * 64; i += 256) {
        int qi = i >> 6, d = i & 63;
        float a = 0.f;
        for (int j = 0; j < SS; ++j) a += P[qi * SS + j] * Vs[j * 64 + d];
        out[((size_t)bb * SS + qi) * 512 + h * 64 + d] = f2bf(a);
    }
}

// ---------------------------------------------------------------------------
// Temporal attention: S=100, one block per (batch, head).
// K/V in LDS (51.2 KB); one thread per query row, two-pass softmax.
// ---------------------------------------------------------------------------
#define ST 100
__global__ __launch_bounds__(128) void attn_temporal(
    const float* __restrict__ qkv, unsigned short* __restrict__ out)
{
    __shared__ float Ks[ST * 64], Vs[ST * 64];
    const int tid = threadIdx.x;
    const int bb  = blockIdx.x >> 3;
    const int h   = blockIdx.x & 7;
    const float* base = qkv + (size_t)bb * ST * 1536 + h * 64;

    for (int i = tid; i < ST * 64; i += 128) {
        int s = i >> 6, d = i & 63;
        Ks[i] = base[(size_t)s * 1536 + 512 + d];
        Vs[i] = base[(size_t)s * 1536 + 1024 + d];
    }
    __syncthreads();

    if (tid < ST) {
        float q[64];
        const float* qp = base + (size_t)tid * 1536;
        #pragma unroll
        for (int d = 0; d < 64; ++d) q[d] = qp[d] * 0.125f;

        float m = -1e30f;
        for (int j = 0; j < ST; ++j) {
            float s = 0.f;
            #pragma unroll
            for (int d = 0; d < 64; ++d) s += q[d] * Ks[j * 64 + d];
            m = fmaxf(m, s);
        }
        float o[64];
        #pragma unroll
        for (int d = 0; d < 64; ++d) o[d] = 0.f;
        float l = 0.f;
        for (int j = 0; j < ST; ++j) {
            float s = 0.f;
            #pragma unroll
            for (int d = 0; d < 64; ++d) s += q[d] * Ks[j * 64 + d];
            float e = __expf(s - m);
            l += e;
            #pragma unroll
            for (int d = 0; d < 64; ++d) o[d] += e * Vs[j * 64 + d];
        }
        float inv = 1.f / l;
        unsigned short* op = out + ((size_t)bb * ST + tid) * 512 + h * 64;
        #pragma unroll
        for (int d = 0; d < 64; ++d) op[d] = f2bf(o[d] * inv);
    }
}

// ---------------------------------------------------------------------------
// Host orchestration
// ---------------------------------------------------------------------------
extern "C" void kernel_launch(void* const* d_in, const int* in_sizes, int n_in,
                              void* d_out, int out_size, void* d_ws, size_t ws_size,
                              hipStream_t stream) {
    (void)in_sizes; (void)n_in; (void)out_size; (void)ws_size;
    const int Bb = 8, Tt = 100, Nn = 25, Dm = 512, DFFn = 2048, DEPTHn = 4;
    const int TOK = Bb * Tt * Nn;              // 20000
    const size_t SZ = (size_t)TOK * Dm;        // elements per [TOK,512] tensor

    const float* x        = (const float*)d_in[0];
    const float* sp_ln_g  = (const float*)d_in[1];
    const float* sp_ln_b  = (const float*)d_in[2];
    const float* sp_qkv_w = (const float*)d_in[3];
    const float* sp_proj_w= (const float*)d_in[4];
    const float* sp_proj_b= (const float*)d_in[5];
    const float* tp_ln_g  = (const float*)d_in[6];
    const float* tp_ln_b  = (const float*)d_in[7];
    const float* tp_qkv_w = (const float*)d_in[8];
    const float* tp_proj_w= (const float*)d_in[9];
    const float* tp_proj_b= (const float*)d_in[10];
    const float* ff_ln_g  = (const float*)d_in[11];
    const float* ff_ln_b  = (const float*)d_in[12];
    const float* ff_w1    = (const float*)d_in[13];
    const float* ff_b1    = (const float*)d_in[14];
    const float* ff_w2    = (const float*)d_in[15];
    const float* ff_b2    = (const float*)d_in[16];

    // ---- workspace layout ----
    char* p = (char*)d_ws;
    float* bufX = (float*)p; p += SZ * 4;                  // current x (f32)
    float* T1   = (float*)p; p += SZ * 4;                  // LN out f32 / zn
    float* T3   = (float*)p; p += SZ * 4;                  // y / z (f32)
    float* T4   = (float*)p; p += SZ * 4;                  // yt / x_new (f32)
    float* Bigf = (float*)p; p += (size_t)TOK * 1536 * 4;  // qkv (f32)
    unsigned short* T1b  = (unsigned short*)p; p += SZ * 2;            // LN out bf16
    unsigned short* T2b  = (unsigned short*)p; p += SZ * 2;            // attn out bf16
    unsigned short* Bigb = (unsigned short*)p; p += (size_t)TOK * 2048 * 2; // FF hidden bf16
    unsigned short* Wb   = (unsigned short*)p;                          // packed weights

    const size_t wQKV = (size_t)Dm * 3 * Dm;   // 786432
    const size_t wPRJ = (size_t)Dm * Dm;       // 262144
    const size_t wFF  = (size_t)Dm * DFFn;     // 1048576
    const size_t perL = 2 * wQKV + 2 * wPRJ + 2 * wFF;

    const dim3 blk(256);

    // ---- one-time weight pack: f32 [K,N] -> bf16 [N,K] ----
    for (int l = 0; l < DEPTHn; ++l) {
        unsigned short* wl = Wb + (size_t)l * perL;
        pack_tr<<<(int)((wQKV + 255) / 256), blk, 0, stream>>>(
            sp_qkv_w + (size_t)l * wQKV, wl, Dm, 3 * Dm);
        pack_tr<<<(int)((wPRJ + 255) / 256), blk, 0, stream>>>(
            sp_proj_w + (size_t)l * wPRJ, wl + wQKV, Dm, Dm);
        pack_tr<<<(int)((wQKV + 255) / 256), blk, 0, stream>>>(
            tp_qkv_w + (size_t)l * wQKV, wl + wQKV + wPRJ, Dm, 3 * Dm);
        pack_tr<<<(int)((wPRJ + 255) / 256), blk, 0, stream>>>(
            tp_proj_w + (size_t)l * wPRJ, wl + 2 * wQKV + wPRJ, Dm, Dm);
        pack_tr<<<(int)((wFF + 255) / 256), blk, 0, stream>>>(
            ff_w1 + (size_t)l * wFF, wl + 2 * wQKV + 2 * wPRJ, Dm, DFFn);
        pack_tr<<<(int)((wFF + 255) / 256), blk, 0, stream>>>(
            ff_w2 + (size_t)l * wFF, wl + 2 * wQKV + 2 * wPRJ + wFF, DFFn, Dm);
    }

    hipMemcpyAsync(bufX, x, SZ * sizeof(float), hipMemcpyDeviceToDevice, stream);

    const int gM = (TOK + BM - 1) / BM;        // 157
    const size_t trTotal = SZ / 4;             // float4 elements
    const int trBlocks = (int)((trTotal + 255) / 256);

    for (int l = 0; l < DEPTHn; ++l) {
        unsigned short* wl   = Wb + (size_t)l * perL;
        unsigned short* sqwT = wl;
        unsigned short* spwT = wl + wQKV;
        unsigned short* tqwT = wl + wQKV + wPRJ;
        unsigned short* tpwT = wl + 2 * wQKV + wPRJ;
        unsigned short* w1T  = wl + 2 * wQKV + 2 * wPRJ;
        unsigned short* w2T  = wl + 2 * wQKV + 2 * wPRJ + wFF;
        const float* spb = sp_proj_b + (size_t)l * Dm;
        const float* tpb = tp_proj_b + (size_t)l * Dm;
        const float* b1  = ff_b1 + (size_t)l * DFFn;
        const float* b2  = ff_b2 + (size_t)l * Dm;

        // ---- spatial attention block ----
        ln_kernel<<<TOK, blk, 0, stream>>>(bufX, sp_ln_g + l * Dm, sp_ln_b + l * Dm, T1, T1b);
        gemm_bf16<<<dim3(3 * Dm / BN, gM), blk, 0, stream>>>(
            T1b, sqwT, nullptr, nullptr, nullptr, Bigf, nullptr, TOK, 3 * Dm, Dm, 0);
        attn_spatial<<<Bb * Tt * 8, blk, 0, stream>>>(Bigf, T2b);
        gemm_bf16<<<dim3(Dm / BN, gM), blk, 0, stream>>>(
            T2b, spwT, spb, bufX, nullptr, T3, nullptr, TOK, Dm, Dm, 0);  // y = attn + x

        // ---- [B,T,N,D] -> [B,N,T,D] ----
        transpose_kernel<<<trBlocks, blk, 0, stream>>>(
            (const float4*)T3, (float4*)T4, Tt, Nn, trTotal);

        // ---- temporal attention block ----
        ln_kernel<<<TOK, blk, 0, stream>>>(T4, tp_ln_g + l * Dm, tp_ln_b + l * Dm, T1, T1b);
        gemm_bf16<<<dim3(3 * Dm / BN, gM), blk, 0, stream>>>(
            T1b, tqwT, nullptr, nullptr, nullptr, Bigf, nullptr, TOK, 3 * Dm, Dm, 0);
        attn_temporal<<<Bb * Nn * 8, dim3(128), 0, stream>>>(Bigf, T2b);
        gemm_bf16<<<dim3(Dm / BN, gM), blk, 0, stream>>>(
            T2b, tpwT, tpb, T4, nullptr, T3, nullptr, TOK, Dm, Dm, 0);    // z = attn + yt

        // ---- feed-forward: x_new = zn + gelu(zn@w1+b1)@w2+b2 + z ----
        ln_kernel<<<TOK, blk, 0, stream>>>(T3, ff_ln_g + l * Dm, ff_ln_b + l * Dm, T1, T1b); // zn
        gemm_bf16<<<dim3(DFFn / BN, gM), blk, 0, stream>>>(
            T1b, w1T, b1, nullptr, nullptr, nullptr, Bigb, TOK, DFFn, Dm, 1);  // GELU, bf16 out
        gemm_bf16<<<dim3(Dm / BN, gM), blk, 0, stream>>>(
            Bigb, w2T, b2, T1, T3, T4, nullptr, TOK, Dm, DFFn, 0);             // + zn + z

        // ---- [B,N,T,D] -> [B,T,N,D] ----
        transpose_kernel<<<trBlocks, blk, 0, stream>>>(
            (const float4*)T4, (float4*)bufX, Nn, Tt, trTotal);
    }

    hipMemcpyAsync(d_out, bufX, SZ * sizeof(float), hipMemcpyDeviceToDevice, stream);
}
